// TemporalSelfAttention_14465449853378
// MI455X (gfx1250) — matrix operand order repaired and checked by reading, not compiled, not run on hardware
//
#include <hip/hip_runtime.h>
#include <hip/hip_bf16.h>

typedef __attribute__((ext_vector_type(16))) _Float16 v16h;
typedef __attribute__((ext_vector_type(8)))  float    v8f;
typedef unsigned int v4u __attribute__((ext_vector_type(4)));
typedef int          v8i __attribute__((ext_vector_type(8)));
typedef int          v4i __attribute__((ext_vector_type(4)));

#if __has_builtin(__builtin_amdgcn_tensor_load_to_lds)
#define HAVE_TDM 1
#endif

// LDS pitches
#define LDA 72   // halfs per A row (64 + 8 pad, 144B = 16B aligned)
#define LDW 68   // floats per W row (64 + 4 pad via TDM pad feature, 272B)

// ---------------- TDM: load 64(Nrows) x 64(Kcols) fp32 weight tile ----------
// Row stride in global = strideK elements. LDS gets pitch LDW floats via the
// TDM pad feature (pad_interval=64 dwords -> code 5, pad_amount=4 dwords ->
// code 3), which both breaks the power-of-two bank stride and keeps every row
// 16B-aligned so fragment reads merge into ds_load_b128.
__device__ __forceinline__ void tdm_load_w64x64(const float* gsrc,
                                                void* lds_dst, int strideK) {
#if defined(HAVE_TDM)
  unsigned long long ga = (unsigned long long)(uintptr_t)gsrc;
  unsigned int lds = (unsigned int)(uintptr_t)lds_dst;
  v4u g0;
  g0[0] = 1u;                                          // count=1 (valid D#)
  g0[1] = lds;                                         // lds_addr (bytes)
  g0[2] = (unsigned int)(ga & 0xFFFFFFFFu);            // global_addr[31:0]
  g0[3] = (unsigned int)((ga >> 32) & 0x01FFFFFFu)     // global_addr[56:32]
        | 0x80000000u;                                 // type=2 ("image")
  v8i g1;
  g1[0] = (2 << 16)        // data_size = 4B
        | (1 << 20)        // pad_enable
        | (5 << 22)        // pad_interval: every 64 dwords
        | (3 << 25);       // pad_amount: 4 dwords
  g1[1] = (64 << 16);      // tensor_dim0[15:0]  (=64)
  g1[2] = (64 << 16);      // tensor_dim0 hi=0 | tensor_dim1[15:0] (=64)
  g1[3] = (64 << 16);      // tensor_dim1 hi=0 | tile_dim0 = 64
  g1[4] = 64;              // tile_dim1 = 64, tile_dim2 = 0
  g1[5] = strideK;         // tensor_dim0_stride[31:0]
  g1[6] = 0;
  g1[7] = 0;
  v4i z4 = {};
  v8i z8 = {};
  // 6-arg form (clang-23 / therock-10.0 headers): groups 2,3 + extra group,
  // then cache policy.
  __builtin_amdgcn_tensor_load_to_lds(g0, g1, z4, z4, z8, 0);
#endif
}

__device__ __forceinline__ void tdm_wait() {
#if defined(HAVE_TDM)
  __builtin_amdgcn_s_wait_tensorcnt(0);
#endif
}

// ---------------- WMMA fragment helpers (wave32, 16x16x32 f16 -> f32) -------
// A (16x32 f16): lanes 0-15: M=lane, K=kb+0..7 (VGPR0-3), kb+16..23 (VGPR4-7);
// lanes 16-31: M=lane-16, kb+8..15 / kb+24..31.
__device__ __forceinline__ v16h frag_a(const _Float16* As, int row0, int ks) {
  const int lane = threadIdx.x & 31;
  const int m  = row0 + (lane & 15);
  const int kb = ks + ((lane >> 4) << 3);
  v16h a;
#pragma unroll
  for (int i = 0; i < 4; ++i) {
    a[2*i]     = As[m*LDA + kb + 2*i];
    a[2*i + 1] = As[m*LDA + kb + 2*i + 1];
    a[8 + 2*i]     = As[m*LDA + kb + 16 + 2*i];
    a[8 + 2*i + 1] = As[m*LDA + kb + 16 + 2*i + 1];
  }
  return a;
}

// B (32x16 f16): lanes 0-15: N=lane, K=ks+0..7, ks+16..23; lanes 16-31:
// N=lane-16, K=ks+8..15, ks+24..31. Weight tile is stored N-major fp32 in
// LDS -> two runs of 8 floats per lane (2x2 ds_load_b128) + cvt to f16.
__device__ __forceinline__ v16h frag_b_w(const float* Ws, int col0, int ks) {
  const int lane = threadIdx.x & 31;
  const int n  = col0 + (lane & 15);
  const int kb = ks + ((lane >> 4) << 3);
  const float* src = Ws + n*LDW + kb;
  v16h b;
#pragma unroll
  for (int c = 0; c < 16; ++c)
    b[c] = (_Float16)src[c < 8 ? c : c + 8];
  return b;
}

__device__ __forceinline__ v8f wmma_f16(v16h a, v16h b, v8f c) {
  return __builtin_amdgcn_wmma_f32_16x16x32_f16(false, a, false, b, (short)0, c,
                                                false, false);
}

// D 16x16 f32: lane: N=lane&15, M = 8*(lane>>4) + vgpr_index

// ---------------- Kernel 1: v = value @ Wv^T + bv, scatter to (bq,h,y,x,d) --
// M=80000 (bq*40000+pix), N=256 (h*32+d), K=256. Block 64x64, 4 waves.
__global__ void __launch_bounds__(128)
k1_value_proj(const float* __restrict__ value, const float* __restrict__ Wv,
              const float* __restrict__ bv, float* __restrict__ v_img)
{
  constexpr int K = 256;
  __shared__ _Float16 As[64 * LDA];
  __shared__ float    Ws[64 * LDW];
  const int tid  = threadIdx.x;
  const int wave = tid >> 5;
  const int lane = tid & 31;
  const int m0 = blockIdx.x * 64;
  const int n0 = blockIdx.y * 64;
  const float* wsrc = Wv + (size_t)n0 * K;

  v8f acc[4] = {};
  for (int k0 = 0; k0 < K; k0 += 64) {
    if (tid < 32) tdm_load_w64x64(wsrc + k0, Ws, K);   // async weight tile
#if !defined(HAVE_TDM)
#pragma unroll
    for (int i = 0; i < 32; ++i) {
      int idx = tid + i*128, nn = idx >> 6, kk = idx & 63;
      Ws[nn*LDW + kk] = wsrc[(size_t)nn*K + k0 + kk];
    }
#endif
    if (k0 + 64 < K) __builtin_prefetch(&value[(size_t)m0*K + k0 + 64], 0, 1);
#pragma unroll
    for (int i = 0; i < 32; ++i) {           // A: 64x64 fp32 -> f16
      int idx = tid + i*128, r = idx >> 6, c = idx & 63;
      As[r*LDA + c] = (_Float16)value[(size_t)(m0 + r)*K + k0 + c];
    }
    tdm_wait();
    __syncthreads();
#pragma unroll
    for (int ks = 0; ks < 64; ks += 32) {
      v16h a = frag_a(As, wave*16, ks);
#pragma unroll
      for (int nb = 0; nb < 4; ++nb)
        acc[nb] = wmma_f16(a, frag_b_w(Ws, nb*16, ks), acc[nb]);
    }
    __syncthreads();
  }

  const int bq = m0 / 40000;                 // block fully inside one bq
  const int nlane = lane & 15, mbase = (lane >> 4) << 3;
#pragma unroll
  for (int nb = 0; nb < 4; ++nb) {
    int col = n0 + nb*16 + nlane;            // = h*32 + d
    int h = col >> 5, d = col & 31;
    float bias = bv[col];
#pragma unroll
    for (int i = 0; i < 8; ++i) {
      int row = m0 + wave*16 + mbase + i;
      int pix = row - bq*40000;
      v_img[((size_t)(bq*8 + h)*40000 + pix)*32 + d] = acc[nb][i] + bias;
    }
  }
}

// ---------------- Kernel 2: so/aw = [value0 || query] @ [Wso^T || Waw^T] ----
// M=40000, N=192 (0..127 so+bso, 128..191 aw+baw), K=512.
__global__ void __launch_bounds__(128)
k2_offsets(const float* __restrict__ value0, const float* __restrict__ query,
           const float* __restrict__ Wso, const float* __restrict__ Waw,
           const float* __restrict__ bso, const float* __restrict__ baw,
           float* __restrict__ so_aw)
{
  constexpr int K = 512;
  __shared__ _Float16 As[64 * LDA];
  __shared__ float    Ws[64 * LDW];
  const int tid  = threadIdx.x;
  const int wave = tid >> 5;
  const int lane = tid & 31;
  const int m0 = blockIdx.x * 64;
  const int n0 = blockIdx.y * 64;
  // n-blocks 0,1 are Wso rows; block 2 is Waw rows (128 boundary = 2*64)
  const float* wsrc = (n0 < 128) ? (Wso + (size_t)n0 * K)
                                 : (Waw + (size_t)(n0 - 128) * K);

  v8f acc[4] = {};
  for (int k0 = 0; k0 < K; k0 += 64) {
    if (tid < 32) tdm_load_w64x64(wsrc + k0, Ws, K);
#if !defined(HAVE_TDM)
#pragma unroll
    for (int i = 0; i < 32; ++i) {
      int idx = tid + i*128, nn = idx >> 6, kk = idx & 63;
      Ws[nn*LDW + kk] = wsrc[(size_t)nn*K + k0 + kk];
    }
#endif
    const float* __restrict__ Asrc = (k0 < 256) ? value0 : query;
    const int kofs = (k0 < 256) ? k0 : (k0 - 256);
#pragma unroll
    for (int i = 0; i < 32; ++i) {
      int idx = tid + i*128, r = idx >> 6, c = idx & 63;
      As[r*LDA + c] = (_Float16)Asrc[(size_t)(m0 + r)*256 + kofs + c];
    }
    tdm_wait();
    __syncthreads();
#pragma unroll
    for (int ks = 0; ks < 64; ks += 32) {
      v16h a = frag_a(As, wave*16, ks);
#pragma unroll
      for (int nb = 0; nb < 4; ++nb)
        acc[nb] = wmma_f16(a, frag_b_w(Ws, nb*16, ks), acc[nb]);
    }
    __syncthreads();
  }

  const int nlane = lane & 15, mbase = (lane >> 4) << 3;
#pragma unroll
  for (int nb = 0; nb < 4; ++nb) {
    int col = n0 + nb*16 + nlane;
    float bias = (col < 128) ? bso[col] : baw[col - 128];
#pragma unroll
    for (int i = 0; i < 8; ++i) {
      int row = m0 + wave*16 + mbase + i;
      so_aw[(size_t)row*192 + col] = acc[nb][i] + bias;
    }
  }
}

// ---------------- Kernel 3: softmax + bilinear sampling + mean over NBQ -----
// thread = (q, h); v_img layout (bq, h, y, x, d), d=32 contiguous (128B).
__global__ void __launch_bounds__(256)
k3_sample(const float* __restrict__ v_img, const float* __restrict__ so_aw,
          const float* __restrict__ ref, float* __restrict__ attn)
{
  const int t = blockIdx.x * blockDim.x + threadIdx.x;
  if (t >= 40000 * 8) return;
  const int h = t & 7;
  const int q = t >> 3;

  float4 acc[8];
#pragma unroll
  for (int i = 0; i < 8; ++i) acc[i] = make_float4(0.f, 0.f, 0.f, 0.f);

#pragma unroll
  for (int bq = 0; bq < 2; ++bq) {
    const float* so  = so_aw + (size_t)q*192 + (h*2 + bq)*8;
    const float* awp = so_aw + (size_t)q*192 + 128 + (h*2 + bq)*4;
    float l0 = awp[0], l1 = awp[1], l2 = awp[2], l3 = awp[3];
    float mx = fmaxf(fmaxf(l0, l1), fmaxf(l2, l3));
    float e0 = __expf(l0 - mx), e1 = __expf(l1 - mx);
    float e2 = __expf(l2 - mx), e3 = __expf(l3 - mx);
    float inv = 1.f / (e0 + e1 + e2 + e3);
    float sm[4] = { e0*inv, e1*inv, e2*inv, e3*inv };

    const float refx = ref[((size_t)bq*40000 + q)*2 + 0];
    const float refy = ref[((size_t)bq*40000 + q)*2 + 1];
    const float* img = v_img + (size_t)(bq*8 + h)*40000*32;

#pragma unroll
    for (int p = 0; p < 4; ++p) {
      float wgt = sm[p];
      float gx = refx*200.f + so[2*p]     - 0.5f;   // loc*W - 0.5
      float gy = refy*200.f + so[2*p + 1] - 0.5f;
      float x0f = floorf(gx), y0f = floorf(gy);
      int   x0 = (int)x0f,    y0 = (int)y0f;
      float wx1 = gx - x0f, wx0 = 1.f - wx1;
      float wy1 = gy - y0f, wy0 = 1.f - wy1;
      const int   xs[2]  = { x0, x0 + 1 };
      const int   ys[2]  = { y0, y0 + 1 };
      const float wxs[2] = { wx0, wx1 };
      const float wys[2] = { wy0, wy1 };
#pragma unroll
      for (int cy = 0; cy < 2; ++cy) {
#pragma unroll
        for (int cx = 0; cx < 2; ++cx) {
          int xi = xs[cx], yi = ys[cy];
          if (xi < 0 || xi >= 200 || yi < 0 || yi >= 200) continue;
          float w = wgt * wxs[cx] * wys[cy];
          const float4* p4 = (const float4*)(img + ((size_t)yi*200 + xi)*32);
#pragma unroll
          for (int i = 0; i < 8; ++i) {
            float4 v = p4[i];
            acc[i].x += w * v.x; acc[i].y += w * v.y;
            acc[i].z += w * v.z; acc[i].w += w * v.w;
          }
        }
      }
    }
  }

  float4* out4 = (float4*)(attn + (size_t)q*256 + h*32);
#pragma unroll
  for (int i = 0; i < 8; ++i) {
    acc[i].x *= 0.5f; acc[i].y *= 0.5f; acc[i].z *= 0.5f; acc[i].w *= 0.5f;
    out4[i] = acc[i];
  }
}

// ---------------- Kernel 4: out = attn @ Wo^T + bo + query ------------------
__global__ void __launch_bounds__(128)
k4_out_proj(const float* __restrict__ attn, const float* __restrict__ Wo,
            const float* __restrict__ bo, const float* __restrict__ query,
            float* __restrict__ out)
{
  constexpr int K = 256;
  __shared__ _Float16 As[64 * LDA];
  __shared__ float    Ws[64 * LDW];
  const int tid  = threadIdx.x;
  const int wave = tid >> 5;
  const int lane = tid & 31;
  const int m0 = blockIdx.x * 64;
  const int n0 = blockIdx.y * 64;
  const float* wsrc = Wo + (size_t)n0 * K;

  v8f acc[4] = {};
  for (int k0 = 0; k0 < K; k0 += 64) {
    if (tid < 32) tdm_load_w64x64(wsrc + k0, Ws, K);
#if !defined(HAVE_TDM)
#pragma unroll
    for (int i = 0; i < 32; ++i) {
      int idx = tid + i*128, nn = idx >> 6, kk = idx & 63;
      Ws[nn*LDW + kk] = wsrc[(size_t)nn*K + k0 + kk];
    }
#endif
    if (k0 + 64 < K) __builtin_prefetch(&attn[(size_t)m0*K + k0 + 64], 0, 1);
#pragma unroll
    for (int i = 0; i < 32; ++i) {
      int idx = tid + i*128, r = idx >> 6, c = idx & 63;
      As[r*LDA + c] = (_Float16)attn[(size_t)(m0 + r)*K + k0 + c];
    }
    tdm_wait();
    __syncthreads();
#pragma unroll
    for (int ks = 0; ks < 64; ks += 32) {
      v16h a = frag_a(As, wave*16, ks);
#pragma unroll
      for (int nb = 0; nb < 4; ++nb)
        acc[nb] = wmma_f16(a, frag_b_w(Ws, nb*16, ks), acc[nb]);
    }
    __syncthreads();
  }

  const int nlane = lane & 15, mbase = (lane >> 4) << 3;
#pragma unroll
  for (int nb = 0; nb < 4; ++nb) {
    int col = n0 + nb*16 + nlane;
    float bias = bo[col];
#pragma unroll
    for (int i = 0; i < 8; ++i) {
      int row = m0 + wave*16 + mbase + i;
      out[(size_t)row*256 + col] = acc[nb][i] + bias + query[(size_t)row*256 + col];
    }
  }
}

// ---------------------------------------------------------------------------
extern "C" void kernel_launch(void* const* d_in, const int* in_sizes, int n_in,
                              void* d_out, int out_size, void* d_ws, size_t ws_size,
                              hipStream_t stream) {
  const float* query = (const float*)d_in[0];   // (1, 40000, 256)
  const float* value = (const float*)d_in[1];   // (2, 40000, 256)
  const float* refp  = (const float*)d_in[2];   // (2, 40000, 1, 2)
  const float* Wv    = (const float*)d_in[3];   // (256, 256)
  const float* bv    = (const float*)d_in[4];
  const float* Wso   = (const float*)d_in[5];   // (128, 512)
  const float* bso   = (const float*)d_in[6];
  const float* Waw   = (const float*)d_in[7];   // (64, 512)
  const float* baw   = (const float*)d_in[8];
  const float* Wo    = (const float*)d_in[9];   // (256, 256)
  const float* bo    = (const float*)d_in[10];
  float* out = (float*)d_out;

  char* ws = (char*)d_ws;
  float* v_img = (float*)ws;                                 // 81,920,000 B
  float* so_aw = (float*)(ws + (size_t)81920000);            // 30,720,000 B
  float* attn  = (float*)(ws + (size_t)81920000 + 30720000); // 40,960,000 B

  dim3 blk(128);
  k1_value_proj<<<dim3(1250, 4), blk, 0, stream>>>(value, Wv, bv, v_img);
  k2_offsets  <<<dim3(625, 3),  blk, 0, stream>>>(value, query, Wso, Waw, bso, baw, so_aw);
  k3_sample   <<<dim3(1250),    dim3(256), 0, stream>>>(v_img, so_aw, refp, attn);
  k4_out_proj <<<dim3(625, 4),  blk, 0, stream>>>(attn, Wo, bo, query, out);
}
